// MultiHeadSelfAttention_84164179132925
// MI455X (gfx1250) — matrix operand order repaired and checked by reading, not compiled, not run on hardware
//
#include <hip/hip_runtime.h>
#include <hip/hip_bf16.h>
#include <math.h>

typedef __attribute__((ext_vector_type(16))) __bf16 v16bf;
typedef __attribute__((ext_vector_type(8)))  __bf16 v8bf;
typedef __attribute__((ext_vector_type(8)))  float  v8f;

#define BB 2
#define TT 2048
#define DDIM 2048
#define HH 16
#define HDIM 128
#define MROWS (BB*TT)      /* 4096 */
#define NQKV  (3*DDIM)     /* 6144 */

#define AKP 40    /* padded row stride (elems) for 32-K LDS tiles  (80B)  */
#define KTP 136   /* padded row stride (elems) for Kt 128-hd rows  (272B) */
#define VTP 40    /* padded row stride (elems) for Vt 32-key rows  (80B)  */
#define PSP 40    /* padded row stride (elems) for P staging       (80B)  */

__device__ __forceinline__ unsigned short f32_to_bf16(float f) {
    unsigned int u = __float_as_uint(f);
    unsigned int r = u + 0x7FFFu + ((u >> 16) & 1u);
    return (unsigned short)(r >> 16);
}
__device__ __forceinline__ float bf16_to_f32(unsigned short u) {
    return __uint_as_float(((unsigned int)u) << 16);
}

// Async global->LDS copy (gfx1250, ASYNCcnt-tracked).
__device__ __forceinline__ void async_g2l_b128(unsigned ldsoff, const void* g) {
    asm volatile("global_load_async_to_lds_b128 %0, %1, off"
                 :: "v"(ldsoff), "v"((unsigned long long)(size_t)g)
                 : "memory");
}
__device__ __forceinline__ void wait_async0() {
    asm volatile("s_wait_asynccnt 0x0" ::: "memory");
}
// 32B fragment from 16B-aligned LDS (two b128 halves at p and p+16)
__device__ __forceinline__ v16bf ld_frag16(const unsigned short* p) {
    v16bf r;
    ((v8bf*)&r)[0] = *(const v8bf*)(p);
    ((v8bf*)&r)[1] = *(const v8bf*)(p + 16);
    return r;
}

// ---------------- f32 -> bf16 conversion ----------------
__global__ void cvt_f32_bf16(const float* __restrict__ in,
                             unsigned short* __restrict__ out, int n) {
    int i = blockIdx.x * blockDim.x + threadIdx.x;
    if (i < n) out[i] = f32_to_bf16(in[i]);
}

// ---------------- GEMM: C = A * W^T + bias --------------
// Block tile 128x128, BK=32, double-buffered async LDS staging.
// 8 waves; wave (mw,nw) computes 64x32: 4 A-frags x 2 W-frags = 8 WMMAs/K-step.
template <bool OUT_BF16>
__global__ __launch_bounds__(256) void gemm_bf16_wmma(
    const unsigned short* __restrict__ A,
    const unsigned short* __restrict__ W,
    const float* __restrict__ bias,
    void* __restrict__ Cout,
    int M, int N, int K)
{
    __shared__ __align__(32) unsigned short At[2][128 * AKP];  // 2 x 10KB
    __shared__ __align__(32) unsigned short Wt[2][128 * AKP];  // 2 x 10KB

    const int tid  = threadIdx.x;
    const int lane = tid & 31;
    const int wib  = tid >> 5;
    const int lm = lane & 15, lg = lane >> 4;

    const int nBlk = N >> 7;
    const int bm = (int)blockIdx.x / nBlk;
    const int bn = (int)blockIdx.x % nBlk;
    const int m0blk = bm << 7;
    const int n0blk = bn << 7;
    const int m00 = (wib >> 2) << 6;     // 0 or 64
    const int n00 = (wib & 3) << 5;      // 0,32,64,96

    // staging chunks: 512 x 16B per tile, 2 per thread (rows of 4 chunks)
    const int c0 = tid, c1 = tid + 256;
    const int ar0 = c0 >> 2, ak0 = (c0 & 3) << 3;
    const int ar1 = c1 >> 2, ak1 = (c1 & 3) << 3;
    const unsigned atB0 = (unsigned)(size_t)At[0], atB1 = (unsigned)(size_t)At[1];
    const unsigned wtB0 = (unsigned)(size_t)Wt[0], wtB1 = (unsigned)(size_t)Wt[1];

    const unsigned short* Ag = A + (size_t)m0blk * K;
    const unsigned short* Wg = W + (size_t)n0blk * K;

    auto stage = [&](int buf, int k) {
        unsigned atB = buf ? atB1 : atB0;
        unsigned wtB = buf ? wtB1 : wtB0;
        async_g2l_b128(atB + (unsigned)((ar0 * AKP + ak0) << 1),
                       Ag + (size_t)ar0 * K + k + ak0);
        async_g2l_b128(atB + (unsigned)((ar1 * AKP + ak1) << 1),
                       Ag + (size_t)ar1 * K + k + ak1);
        async_g2l_b128(wtB + (unsigned)((ar0 * AKP + ak0) << 1),
                       Wg + (size_t)ar0 * K + k + ak0);
        async_g2l_b128(wtB + (unsigned)((ar1 * AKP + ak1) << 1),
                       Wg + (size_t)ar1 * K + k + ak1);
    };

    v8f acc[4][2] = {};

    stage(0, 0);
    wait_async0();
    __syncthreads();

    int buf = 0;
    for (int k = 0; k < K; k += 32) {
        if (k + 32 < K) stage(buf ^ 1, k + 32);   // prefetch next tile

        const unsigned short* at = At[buf];
        const unsigned short* wt = Wt[buf];
        v16bf af[4];
#pragma unroll
        for (int mi = 0; mi < 4; ++mi) {
            const unsigned short* ap = at + (m00 + mi * 16 + lm) * AKP + (lg << 3);
            ((v8bf*)&af[mi])[0] = *(const v8bf*)(ap);
            ((v8bf*)&af[mi])[1] = *(const v8bf*)(ap + 16);
        }
#pragma unroll
        for (int ci = 0; ci < 2; ++ci) {
            v16bf bf = ld_frag16(wt + (n00 + ci * 16 + lm) * AKP + (lg << 3));
#pragma unroll
            for (int mi = 0; mi < 4; ++mi)
                acc[mi][ci] = __builtin_amdgcn_wmma_f32_16x16x32_bf16(
                    false, af[mi], false, bf, (short)0, acc[mi][ci], false, false);
        }
        wait_async0();      // next tile landed
        __syncthreads();    // everyone done reading current tile
        buf ^= 1;
    }
#pragma unroll
    for (int mi = 0; mi < 4; ++mi) {
#pragma unroll
        for (int ci = 0; ci < 2; ++ci) {
            int n = n0blk + n00 + ci * 16 + lm;
            float bv = bias ? bias[n] : 0.0f;
#pragma unroll
            for (int r = 0; r < 8; ++r) {
                int m = m0blk + m00 + mi * 16 + r + (lg << 3);
                float v = acc[mi][ci][r] + bv;
                if (OUT_BF16)
                    ((unsigned short*)Cout)[(size_t)m * N + n] = f32_to_bf16(v);
                else
                    ((float*)Cout)[(size_t)m * N + n] = v;
            }
        }
    }
}

// -------- RoPE + head split + V transpose ---------------
__global__ void rope_prepare(const unsigned short* __restrict__ qkvb,
                             unsigned short* __restrict__ Qh,
                             unsigned short* __restrict__ Kh,
                             unsigned short* __restrict__ VT)
{
    int i = blockIdx.x * blockDim.x + threadIdx.x;
    if (i >= BB * HH * TT * HDIM) return;
    int hd = i & (HDIM - 1);
    int t  = (i >> 7) & (TT - 1);
    int h  = (i >> 18) & (HH - 1);
    int b  = i >> 22;

    size_t rowbase = (size_t)(b * TT + t) * (size_t)NQKV + (size_t)h * HDIM;
    int ii = hd & 63;
    float ang = (float)t * __expf(-0.14391156831212787f * (float)ii);
    float sn, cs;
    __sincosf(ang, &sn, &cs);
    float qo, ko;
    if (hd < 64) {
        float q1 = bf16_to_f32(qkvb[rowbase + hd]);
        float q2 = bf16_to_f32(qkvb[rowbase + hd + 64]);
        float k1 = bf16_to_f32(qkvb[rowbase + DDIM + hd]);
        float k2 = bf16_to_f32(qkvb[rowbase + DDIM + hd + 64]);
        qo = q1 * cs - q2 * sn;
        ko = k1 * cs - k2 * sn;
    } else {
        float q1 = bf16_to_f32(qkvb[rowbase + hd - 64]);
        float q2 = bf16_to_f32(qkvb[rowbase + hd]);
        float k1 = bf16_to_f32(qkvb[rowbase + DDIM + hd - 64]);
        float k2 = bf16_to_f32(qkvb[rowbase + DDIM + hd]);
        qo = q1 * sn + q2 * cs;
        ko = k1 * sn + k2 * cs;
    }
    float vv = bf16_to_f32(qkvb[rowbase + 2 * DDIM + hd]);

    size_t bh = (size_t)(b * HH + h);
    Qh[(bh * TT + t) * HDIM + hd] = f32_to_bf16(qo);
    Kh[(bh * TT + t) * HDIM + hd] = f32_to_bf16(ko);
    VT[(bh * HDIM + hd) * TT + t] = f32_to_bf16(vv);
}

// ---------------- Flash attention -----------------------
// Block = 8 waves = 128 consecutive query rows of one (b,h).
// K/V tiles (32 keys) staged once per block into LDS via async loads.
__global__ __launch_bounds__(256) void flash_attn(
    const unsigned short* __restrict__ Qh,
    const unsigned short* __restrict__ Kh,
    const unsigned short* __restrict__ VT,
    unsigned short* __restrict__ Ob)   // (B,T,H,HD) bf16 == (B,T,D)
{
    __shared__ __align__(32) unsigned short Kt[32 * KTP];    // [key][hd]
    __shared__ __align__(32) unsigned short Vt[HDIM * VTP];  // [hd][key]
    __shared__ __align__(32) unsigned short Pst[8][16 * PSP];

    const int tid  = threadIdx.x;
    const int lane = tid & 31;
    const int wib  = tid >> 5;
    const int lm = lane & 15, lg = lane >> 4;

    const int blk = blockIdx.x;            // B*H*(T/128) = 512
    const int qg = blk & 15;
    const int h  = (blk >> 4) & (HH - 1);
    const int b  = blk >> 8;
    const int qt = qg * 8 + wib;

    size_t bh = (size_t)(b * HH + h);
    const unsigned short* Qp = Qh + bh * TT * HDIM;
    const unsigned short* Kp = Kh + bh * TT * HDIM;
    const unsigned short* Vp = VT + bh * HDIM * TT;

    v16bf qf[4];
    {
        const unsigned short* qrow = Qp + (size_t)(qt * 16 + lm) * HDIM + (lg << 3);
#pragma unroll
        for (int c = 0; c < 4; ++c) {
            ((v8bf*)&qf[c])[0] = *(const v8bf*)(qrow + c * 32);
            ((v8bf*)&qf[c])[1] = *(const v8bf*)(qrow + c * 32 + 16);
        }
    }
    v8f of[8] = {};
    float mrow[8], lrow[8];
#pragma unroll
    for (int r = 0; r < 8; ++r) { mrow[r] = -1e30f; lrow[r] = 0.0f; }

    const float scale = 0.08838834764831845f;
    const int nblk = ((qg * 128 + 127) >> 5) + 1;
    const int myLastJ = (qt * 16 + 15) >> 5;
    unsigned short* pw = Pst[wib];

    const unsigned ktBase = (unsigned)(size_t)Kt;
    const unsigned vtBase = (unsigned)(size_t)Vt;
    const int kc0 = tid, kc1 = tid + 256;

    for (int j = 0; j < nblk; ++j) {
        const int key0 = j << 5;
        __syncthreads();
        {
            int key, hd0, kl;
            key = kc0 >> 4; hd0 = (kc0 & 15) << 3;
            async_g2l_b128(ktBase + (unsigned)((key * KTP + hd0) << 1),
                           Kp + (size_t)(key0 + key) * HDIM + hd0);
            key = kc1 >> 4; hd0 = (kc1 & 15) << 3;
            async_g2l_b128(ktBase + (unsigned)((key * KTP + hd0) << 1),
                           Kp + (size_t)(key0 + key) * HDIM + hd0);
            hd0 = kc0 >> 2; kl = (kc0 & 3) << 3;
            async_g2l_b128(vtBase + (unsigned)((hd0 * VTP + kl) << 1),
                           Vp + (size_t)hd0 * TT + key0 + kl);
            hd0 = kc1 >> 2; kl = (kc1 & 3) << 3;
            async_g2l_b128(vtBase + (unsigned)((hd0 * VTP + kl) << 1),
                           Vp + (size_t)hd0 * TT + key0 + kl);
        }
        wait_async0();
        __syncthreads();

        if (j > myLastJ) continue;   // wave-uniform

        v8f s[2] = {};
#pragma unroll
        for (int t = 0; t < 2; ++t) {
            const unsigned short* krow = Kt + (t * 16 + lm) * KTP + (lg << 3);
#pragma unroll
            for (int c = 0; c < 4; ++c) {
                v16bf kf = ld_frag16(krow + c * 32);
                s[t] = __builtin_amdgcn_wmma_f32_16x16x32_bf16(
                    false, qf[c], false, kf, (short)0, s[t], false, false);
            }
        }
        float mloc[8];
#pragma unroll
        for (int r = 0; r < 8; ++r) {
            int q = qt * 16 + r + (lg << 3);
            float a0 = s[0][r] * scale;
            float a1 = s[1][r] * scale;
            if (key0 + lm > q)      a0 = -1e30f;
            if (key0 + 16 + lm > q) a1 = -1e30f;
            s[0][r] = a0; s[1][r] = a1;
            mloc[r] = fmaxf(a0, a1);
        }
#pragma unroll
        for (int r = 0; r < 8; ++r) {
            float v = mloc[r];
            v = fmaxf(v, __shfl_xor(v, 1, 32));
            v = fmaxf(v, __shfl_xor(v, 2, 32));
            v = fmaxf(v, __shfl_xor(v, 4, 32));
            v = fmaxf(v, __shfl_xor(v, 8, 32));
            mloc[r] = v;
        }
        float corr[8], ps[8];
#pragma unroll
        for (int r = 0; r < 8; ++r) {
            float mnew = fmaxf(mrow[r], mloc[r]);
            corr[r] = __expf(mrow[r] - mnew);
            mrow[r] = mnew;
            float p0 = __expf(s[0][r] - mnew);
            float p1 = __expf(s[1][r] - mnew);
            int row = r + (lg << 3);
            pw[row * PSP + lm]      = f32_to_bf16(p0);
            pw[row * PSP + 16 + lm] = f32_to_bf16(p1);
            ps[r] = p0 + p1;
        }
#pragma unroll
        for (int r = 0; r < 8; ++r) {
            float v = ps[r];
            v += __shfl_xor(v, 1, 32);
            v += __shfl_xor(v, 2, 32);
            v += __shfl_xor(v, 4, 32);
            v += __shfl_xor(v, 8, 32);
            lrow[r] = lrow[r] * corr[r] + v;
        }
#pragma unroll
        for (int c = 0; c < 8; ++c)
#pragma unroll
            for (int r = 0; r < 8; ++r)
                of[c][r] *= corr[r];

        asm volatile("s_wait_dscnt 0x0" ::: "memory");

        v16bf pf;
        ((v8bf*)&pf)[0] = *(const v8bf*)(pw + lm * PSP + (lg << 3));
        ((v8bf*)&pf)[1] = *(const v8bf*)(pw + lm * PSP + 16 + (lg << 3));

#pragma unroll
        for (int c = 0; c < 8; ++c) {
            v16bf vf = ld_frag16(Vt + (c * 16 + lm) * VTP + (lg << 3));
            of[c] = __builtin_amdgcn_wmma_f32_16x16x32_bf16(
                false, pf, false, vf, (short)0, of[c], false, false);
        }
    }
#pragma unroll
    for (int r = 0; r < 8; ++r) {
        float inv = 1.0f / lrow[r];
        int q = qt * 16 + r + (lg << 3);
        size_t base = (size_t)(b * TT + q) * DDIM + (size_t)h * HDIM;
#pragma unroll
        for (int c = 0; c < 8; ++c)
            Ob[base + c * 16 + lm] = f32_to_bf16(of[c][r] * inv);
    }
}

extern "C" void kernel_launch(void* const* d_in, const int* in_sizes, int n_in,
                              void* d_out, int out_size, void* d_ws, size_t ws_size,
                              hipStream_t stream)
{
    (void)in_sizes; (void)n_in; (void)out_size; (void)ws_size;
    const float* x     = (const float*)d_in[0];
    const float* qkv_w = (const float*)d_in[1];
    const float* qkv_b = (const float*)d_in[2];
    const float* out_w = (const float*)d_in[3];
    const float* out_b = (const float*)d_in[4];
    float* out = (float*)d_out;

    char* ws = (char*)d_ws;
    size_t off = 0;
    auto alloc = [&](size_t bytes) {
        void* p = ws + off;
        off += (bytes + 255) & ~(size_t)255;
        return p;
    };
    unsigned short* xb    = (unsigned short*)alloc((size_t)MROWS * DDIM * 2);
    unsigned short* wqkvb = (unsigned short*)alloc((size_t)NQKV * DDIM * 2);
    unsigned short* wob   = (unsigned short*)alloc((size_t)DDIM * DDIM * 2);
    unsigned short* qkvb  = (unsigned short*)alloc((size_t)MROWS * NQKV * 2);
    unsigned short* QhB   = (unsigned short*)alloc((size_t)BB * HH * TT * HDIM * 2);
    unsigned short* KhB   = (unsigned short*)alloc((size_t)BB * HH * TT * HDIM * 2);
    unsigned short* VTB   = (unsigned short*)alloc((size_t)BB * HH * TT * HDIM * 2);
    unsigned short* ObB   = (unsigned short*)alloc((size_t)MROWS * DDIM * 2);

    int n1 = MROWS * DDIM;
    cvt_f32_bf16<<<(n1 + 255) / 256, 256, 0, stream>>>(x, xb, n1);
    int n2 = NQKV * DDIM;
    cvt_f32_bf16<<<(n2 + 255) / 256, 256, 0, stream>>>(qkv_w, wqkvb, n2);
    int n3 = DDIM * DDIM;
    cvt_f32_bf16<<<(n3 + 255) / 256, 256, 0, stream>>>(out_w, wob, n3);

    {
        int blocks = (MROWS / 128) * (NQKV / 128);   // 32*48 = 1536
        gemm_bf16_wmma<true><<<blocks, 256, 0, stream>>>(
            xb, wqkvb, qkv_b, qkvb, MROWS, NQKV, DDIM);
    }
    {
        int n = BB * HH * TT * HDIM;
        rope_prepare<<<(n + 255) / 256, 256, 0, stream>>>(qkvb, QhB, KhB, VTB);
    }
    {
        int blocks = BB * HH * (TT / 128);   // 512
        flash_attn<<<blocks, 256, 0, stream>>>(QhB, KhB, VTB, ObB);
    }
    {
        int blocks = (MROWS / 128) * (DDIM / 128);   // 32*16 = 512
        gemm_bf16_wmma<false><<<blocks, 256, 0, stream>>>(
            ObB, wob, out_b, out, MROWS, DDIM, DDIM);
    }
}
